// DA_rnn_28741921145345
// MI455X (gfx1250) — matrix operand order repaired and checked by reading, not compiled, not run on hardware
//
#include <hip/hip_runtime.h>
#include <hip/hip_bf16.h>

typedef __attribute__((ext_vector_type(16))) __bf16 v16bf;
typedef __attribute__((ext_vector_type(8)))  __bf16 v8bf;
typedef __attribute__((ext_vector_type(8)))  float  v8f;

#define LD_ACT 2056   // bf16 elems per LDS activation row (1536/2048 used + pad)
#define LD_F   1028   // f32 elems per LDS state row (1024 used + pad)

__device__ __forceinline__ float sigm(float x) { return 1.f / (1.f + __expf(-x)); }

__device__ __forceinline__ v8f wmma_bf16(v16bf a, v16bf b, v8f c) {
  return __builtin_amdgcn_wmma_f32_16x16x32_bf16(false, a, false, b, (short)0, c, false, false);
}

// A fragment (16x32 bf16, M=batch rows): row-major source, this lane's row ptr.
// lane<16: K = {k0..k0+15}; lane>=16: K = {k0+16..k0+31}.
__device__ __forceinline__ v16bf frag_a(const __bf16* __restrict__ row, int k0, int hi) {
  v8bf lo = *(const v8bf*)(row + k0 + hi * 16);
  v8bf hh = *(const v8bf*)(row + k0 + hi * 16 + 8);
  v16bf a;
#pragma unroll
  for (int j = 0; j < 8; ++j) { a[j] = lo[j]; a[8 + j] = hh[j]; }
  return a;
}

// B fragment from fragment-major packed weights: 512 bf16 per fragment, 16 contiguous per lane.
__device__ __forceinline__ v16bf frag_b(const __bf16* __restrict__ pack, long fragIdx, int lane) {
  return *(const v16bf*)(pack + ((fragIdx << 5) + (long)lane) * 16);
}

// ---------------- weight packing: W (N x K) row-major f32 -> fragment-major bf16 ----
// packed[((nt*KT+kt)*32 + lane)*16 + j] = W[nt*16 + (lane&15)][kt*32 + (lane>=16?16:0) + j]
__global__ void pack_b(const float* __restrict__ src, __bf16* __restrict__ dst,
                       int N, int K, int srcStride, int colOff) {
  const int KT = K >> 5;
  const long total = (long)N * K;
  for (long idx = (long)blockIdx.x * blockDim.x + threadIdx.x; idx < total;
       idx += (long)gridDim.x * blockDim.x) {
    long frag = idx >> 9;
    int r = (int)(idx & 511);
    int lane = r >> 4, j = r & 15;
    int nt = (int)(frag / KT), kt = (int)(frag % KT);
    int n = nt * 16 + (lane & 15);
    int k = kt * 32 + ((lane >> 4) << 4) + j;
    dst[idx] = (__bf16)src[(long)n * srcStride + colOff + k];
  }
}

// Concatenated [Wih_e | Whh_e] along K (512 + 1024). K-tiles never straddle the seam.
__global__ void pack_b_cat(const float* __restrict__ A, int Ka,
                           const float* __restrict__ Bm, int Kb,
                           __bf16* __restrict__ dst, int N) {
  const int K = Ka + Kb;
  const int KT = K >> 5;
  const long total = (long)N * K;
  for (long idx = (long)blockIdx.x * blockDim.x + threadIdx.x; idx < total;
       idx += (long)gridDim.x * blockDim.x) {
    long frag = idx >> 9;
    int r = (int)(idx & 511);
    int lane = r >> 4, j = r & 15;
    int nt = (int)(frag / KT), kt = (int)(frag % KT);
    int n = nt * 16 + (lane & 15);
    int k = kt * 32 + ((lane >> 4) << 4) + j;
    float v = (k < Ka) ? A[(long)n * Ka + k] : Bm[(long)n * Kb + (k - Ka)];
    dst[idx] = (__bf16)v;
  }
}

// ---------------- main fused DA-RNN kernel: one block = 16 batch rows ----------------
__global__ __launch_bounds__(256)
void da_rnn_main(const float* __restrict__ X, const float* __restrict__ y_prev,
                 const float* __restrict__ enc_w,
                 const float* __restrict__ bih_e, const float* __restrict__ bhh_e,
                 const float* __restrict__ b1, const float* __restrict__ W2,
                 const float* __restrict__ b2,
                 const float* __restrict__ Wih_d, const float* __restrict__ bih_d,
                 const float* __restrict__ bhh_d,
                 const float* __restrict__ fc_w, const float* __restrict__ fc_b,
                 const float* __restrict__ ff_w, const float* __restrict__ ff_b,
                 const __bf16* __restrict__ wcat, const __bf16* __restrict__ whhd,
                 const __bf16* __restrict__ w1dc, const __bf16* __restrict__ w1x,
                 __bf16* __restrict__ xenc, __bf16* __restrict__ xproj,
                 float* __restrict__ out) {
  extern __shared__ char smem[];
  __bf16* sAct = (__bf16*)smem;                          // [16][LD_ACT]
  float*  sH   = (float*)(smem + 16 * LD_ACT * 2);       // [16][LD_F]
  float*  sC   = sH + 16 * LD_F;                         // [16][LD_F]
  float*  sAux = sC + 16 * LD_F;                         // [16][LD_F] alpha / pre / context

  __shared__ float sWx[32];
  __shared__ float sW2[1024];
  __shared__ float sFc[1025];
  __shared__ float sFf[2048];
  __shared__ float sPart[16][16];
  __shared__ float sScore[16][33];
  __shared__ float sBeta[16][33];
  __shared__ float sYt[16];
  __shared__ float sRed[16];

  const int tid  = threadIdx.x;
  const int lane = tid & 31;
  const int wave = tid >> 5;
  const int l15  = lane & 15;
  const int hi   = lane >> 4;
  const int b0   = blockIdx.x * 16;

  // stage small vectors
  if (tid < 32) sWx[tid] = enc_w[2 * 1024 + tid];
  for (int i = tid; i < 1024; i += 256) sW2[i] = W2[i];
  for (int i = tid; i < 1025; i += 256) sFc[i] = fc_w[i];
  for (int i = tid; i < 2048; i += 256) sFf[i] = ff_w[i];
  for (int i = tid; i < 16 * LD_F; i += 256) { sH[i] = 0.f; sC[i] = 0.f; }
  for (int i = tid; i < 16 * LD_ACT; i += 256) sAct[i] = (__bf16)0.f;
  __syncthreads();

  // ---------- encoder input attention: alpha = softmax_i( sum_t X[b,t,i]*w_x[t] ) ----
  for (int idx = tid; idx < 16 * 512; idx += 256) {
    int m = idx >> 9, i = idx & 511;
    const float* xp = X + ((long)(b0 + m) * 32) * 512 + i;
    float acc = 0.f;
#pragma unroll
    for (int t = 0; t < 32; ++t) acc += xp[t * 512] * sWx[t];
    sAux[m * LD_F + i] = acc;
  }
  __syncthreads();
  {
    int m = tid >> 4, p = tid & 15;
    float mx = -1e30f;
    for (int i = p * 32; i < p * 32 + 32; ++i) mx = fmaxf(mx, sAux[m * LD_F + i]);
    sPart[m][p] = mx;
    __syncthreads();
    if (p == 0) { float v = -1e30f; for (int q = 0; q < 16; ++q) v = fmaxf(v, sPart[m][q]); sRed[m] = v; }
    __syncthreads();
    float rm = sRed[m], s = 0.f;
    for (int i = p * 32; i < p * 32 + 32; ++i) {
      float e = __expf(sAux[m * LD_F + i] - rm);
      sAux[m * LD_F + i] = e; s += e;
    }
    sPart[m][p] = s;
    __syncthreads();
    if (p == 0) { float v = 0.f; for (int q = 0; q < 16; ++q) v += sPart[m][q]; sRed[m] = v; }
    __syncthreads();
    float inv = 1.f / sRed[m];
    for (int i = p * 32; i < p * 32 + 32; ++i) sAux[m * LD_F + i] *= inv;
  }
  __syncthreads();

  // ---------- encoder time loop ----------
  for (int t = 0; t < 32; ++t) {
    // x_tilde = alpha * X[:,t,:]  (bf16 into act cols 0..511)
    for (int idx = tid; idx < 16 * 512; idx += 256) {
      int m = idx >> 9, i = idx & 511;
      float xv = X[((long)(b0 + m) * 32 + t) * 512 + i];
      sAct[m * LD_ACT + i] = (__bf16)(sAux[m * LD_F + i] * xv);
    }
    __syncthreads();
    // gates = act(16x1536) @ Wcat^T(1536x4096); wave owns units [wave*128, +128)
    const __bf16* arow = sAct + l15 * LD_ACT;
    for (int ut = 0; ut < 8; ++ut) {
      int u0 = wave * 128 + ut * 16;
      int nt0 = u0 >> 4;
      v8f aI = {0,0,0,0,0,0,0,0}, aF = {0,0,0,0,0,0,0,0};
      v8f aG = {0,0,0,0,0,0,0,0}, aO = {0,0,0,0,0,0,0,0};
#pragma unroll 4
      for (int kt = 0; kt < 48; ++kt) {
        v16bf a = frag_a(arow, kt * 32, hi);
        v16bf bI = frag_b(wcat, (long)(nt0)       * 48 + kt, lane);
        v16bf bF = frag_b(wcat, (long)(nt0 + 64)  * 48 + kt, lane);
        v16bf bG = frag_b(wcat, (long)(nt0 + 128) * 48 + kt, lane);
        v16bf bO = frag_b(wcat, (long)(nt0 + 192) * 48 + kt, lane);
        aI = wmma_bf16(a, bI, aI);
        aF = wmma_bf16(a, bF, aF);
        aG = wmma_bf16(a, bG, aG);
        aO = wmma_bf16(a, bO, aO);
      }
      int u = u0 + l15;
      float bI = bih_e[u]          + bhh_e[u];
      float bF = bih_e[1024 + u]   + bhh_e[1024 + u];
      float bG = bih_e[2048 + u]   + bhh_e[2048 + u];
      float bO = bih_e[3072 + u]   + bhh_e[3072 + u];
#pragma unroll
      for (int r = 0; r < 8; ++r) {
        int m = r + hi * 8;
        float iv = sigm(aI[r] + bI);
        float fv = sigm(aF[r] + bF);
        float gv = tanhf(aG[r] + bG);
        float ov = sigm(aO[r] + bO);
        float cp = sC[m * LD_F + u];
        float cn = fv * cp + iv * gv;
        float hn = ov * tanhf(cn);
        sC[m * LD_F + u] = cn;
        sH[m * LD_F + u] = hn;
        xenc[((long)(b0 + m) * 32 + t) * 1024 + u] = (__bf16)hn;
      }
    }
    __syncthreads();
    // refresh bf16 h copy (act cols 512..1535)
    for (int idx = tid; idx < 16 * 1024; idx += 256) {
      int m = idx >> 10, u = idx & 1023;
      sAct[m * LD_ACT + 512 + u] = (__bf16)sH[m * LD_F + u];
    }
  }
  __threadfence();
  __syncthreads();

  // ---------- Xenc_proj = X_enc @ W1_x^T + b1  (per time slice, WMMA) ----------
  for (int tt = 0; tt < 32; ++tt) {
    const __bf16* arow = xenc + ((long)(b0 + l15) * 32 + tt) * 1024;
    for (int nt8 = 0; nt8 < 8; ++nt8) {
      int f0 = wave * 128 + nt8 * 16;
      int nt = f0 >> 4;
      v8f acc = {0,0,0,0,0,0,0,0};
#pragma unroll 4
      for (int kt = 0; kt < 32; ++kt) {
        v16bf a = frag_a(arow, kt * 32, hi);
        v16bf b = frag_b(w1x, (long)nt * 32 + kt, lane);
        acc = wmma_bf16(a, b, acc);
      }
      int f = f0 + l15;
      float bias = b1[f];
#pragma unroll
      for (int r = 0; r < 8; ++r) {
        int m = r + hi * 8;
        xproj[((long)(b0 + m) * 32 + tt) * 1024 + f] = (__bf16)(acc[r] + bias);
      }
    }
  }
  __threadfence();
  __syncthreads();

  // ---------- decoder init ----------
  for (int i = tid; i < 16 * LD_F; i += 256) { sH[i] = 0.f; sC[i] = 0.f; }
  for (int i = tid; i < 16 * LD_ACT; i += 256) sAct[i] = (__bf16)0.f;
  __syncthreads();

  // ---------- decoder time loop ----------
  for (int t = 0; t < 32; ++t) {
    // 1) pre = [d|c](16x2048) @ W1_dc^T -> sAux (f32)
    {
      const __bf16* arow = sAct + l15 * LD_ACT;
      for (int nt8 = 0; nt8 < 8; ++nt8) {
        int f0 = wave * 128 + nt8 * 16;
        int nt = f0 >> 4;
        v8f acc = {0,0,0,0,0,0,0,0};
#pragma unroll 4
        for (int kt = 0; kt < 64; ++kt) {
          v16bf a = frag_a(arow, kt * 32, hi);
          v16bf b = frag_b(w1dc, (long)nt * 64 + kt, lane);
          acc = wmma_bf16(a, b, acc);
        }
        int f = f0 + l15;
#pragma unroll
        for (int r = 0; r < 8; ++r) { int m = r + hi * 8; sAux[m * LD_F + f] = acc[r]; }
      }
    }
    __syncthreads();
    // 2) scores[b,tt] = w2 . tanh(pre[b] + Xenc_proj[b,tt]) + b2
    for (int p = tid; p < 512; p += 256) {
      int m = p >> 5, tt = p & 31;
      const __bf16* xp = xproj + ((long)(b0 + m) * 32 + tt) * 1024;
      const float* pre = sAux + m * LD_F;
      float s = 0.f;
      for (int e = 0; e < 1024; ++e) s += sW2[e] * tanhf(pre[e] + (float)xp[e]);
      sScore[m][tt] = s + b2[0];
    }
    __syncthreads();
    // 3) beta = softmax_tt(scores)
    if (tid < 16) {
      int m = tid;
      float mx = -1e30f;
      for (int tt = 0; tt < 32; ++tt) mx = fmaxf(mx, sScore[m][tt]);
      float s = 0.f;
      for (int tt = 0; tt < 32; ++tt) { float e = __expf(sScore[m][tt] - mx); sBeta[m][tt] = e; s += e; }
      float inv = 1.f / s;
      for (int tt = 0; tt < 32; ++tt) sBeta[m][tt] *= inv;
    }
    __syncthreads();
    // 4) context[b,e] = sum_tt beta * X_enc  -> sAux (overwrites pre)
    for (int idx = tid; idx < 16 * 1024; idx += 256) {
      int m = idx >> 10, e = idx & 1023;
      const __bf16* xe = xenc + ((long)(b0 + m) * 32) * 1024 + e;
      float acc = 0.f;
#pragma unroll
      for (int tt = 0; tt < 32; ++tt) acc += sBeta[m][tt] * (float)xe[tt * 1024];
      sAux[m * LD_F + e] = acc;
    }
    __syncthreads();
    // 5) y_tilde = context . fc_w[:E] + y_prev[:,t]*fc_w[E] + fc_b
    {
      int m = tid >> 4, p = tid & 15;
      float s = 0.f;
      for (int e = p * 64; e < p * 64 + 64; ++e) s += sAux[m * LD_F + e] * sFc[e];
      sPart[m][p] = s;
    }
    __syncthreads();
    if (tid < 16) {
      float s = 0.f;
      for (int q = 0; q < 16; ++q) s += sPart[tid][q];
      sYt[tid] = s + y_prev[(long)(b0 + tid) * 32 + t] * sFc[1024] + fc_b[0];
    }
    __syncthreads();
    // 6) decoder LSTM: gates = d @ Whh_d^T + y_tilde*Wih_d + biases
    {
      const __bf16* arow = sAct + l15 * LD_ACT;  // d in cols 0..1023
      for (int ut = 0; ut < 8; ++ut) {
        int u0 = wave * 128 + ut * 16;
        int nt0 = u0 >> 4;
        v8f aI = {0,0,0,0,0,0,0,0}, aF = {0,0,0,0,0,0,0,0};
        v8f aG = {0,0,0,0,0,0,0,0}, aO = {0,0,0,0,0,0,0,0};
#pragma unroll 4
        for (int kt = 0; kt < 32; ++kt) {
          v16bf a = frag_a(arow, kt * 32, hi);
          v16bf bI = frag_b(whhd, (long)(nt0)       * 32 + kt, lane);
          v16bf bF = frag_b(whhd, (long)(nt0 + 64)  * 32 + kt, lane);
          v16bf bG = frag_b(whhd, (long)(nt0 + 128) * 32 + kt, lane);
          v16bf bO = frag_b(whhd, (long)(nt0 + 192) * 32 + kt, lane);
          aI = wmma_bf16(a, bI, aI);
          aF = wmma_bf16(a, bF, aF);
          aG = wmma_bf16(a, bG, aG);
          aO = wmma_bf16(a, bO, aO);
        }
        int u = u0 + l15;
        float wI = Wih_d[u],        wF = Wih_d[1024 + u];
        float wG = Wih_d[2048 + u], wO = Wih_d[3072 + u];
        float bI = bih_d[u]        + bhh_d[u];
        float bF = bih_d[1024 + u] + bhh_d[1024 + u];
        float bG = bih_d[2048 + u] + bhh_d[2048 + u];
        float bO = bih_d[3072 + u] + bhh_d[3072 + u];
#pragma unroll
        for (int r = 0; r < 8; ++r) {
          int m = r + hi * 8;
          float yt = sYt[m];
          float iv = sigm(aI[r] + yt * wI + bI);
          float fv = sigm(aF[r] + yt * wF + bF);
          float gv = tanhf(aG[r] + yt * wG + bG);
          float ov = sigm(aO[r] + yt * wO + bO);
          float cp = sC[m * LD_F + u];
          float cn = fv * cp + iv * gv;
          float hn = ov * tanhf(cn);
          sC[m * LD_F + u] = cn;
          sH[m * LD_F + u] = hn;
        }
      }
    }
    __syncthreads();
    // 7) refresh bf16 [d|c] for next step's GEMMs
    for (int idx = tid; idx < 16 * 1024; idx += 256) {
      int m = idx >> 10, u = idx & 1023;
      sAct[m * LD_ACT + u]        = (__bf16)sH[m * LD_F + u];
      sAct[m * LD_ACT + 1024 + u] = (__bf16)sC[m * LD_F + u];
    }
    __syncthreads();
  }

  // ---------- final: y = [d | context] . ff_w + ff_b ----------
  {
    int m = tid >> 4, p = tid & 15;
    float s = 0.f;
    for (int e = p * 64; e < p * 64 + 64; ++e)
      s += sH[m * LD_F + e] * sFf[e] + sAux[m * LD_F + e] * sFf[1024 + e];
    sPart[m][p] = s;
  }
  __syncthreads();
  if (tid < 16) {
    float s = 0.f;
    for (int q = 0; q < 16; ++q) s += sPart[tid][q];
    out[b0 + tid] = s + ff_b[0];
  }
}

// ---------------- launch ----------------
extern "C" void kernel_launch(void* const* d_in, const int* in_sizes, int n_in,
                              void* d_out, int out_size, void* d_ws, size_t ws_size,
                              hipStream_t stream) {
  const float* X      = (const float*)d_in[0];
  const float* y_prev = (const float*)d_in[1];
  const float* enc_w  = (const float*)d_in[2];
  // d_in[3] enc_attn_b: per-row constant inside softmax -> cancels, unused
  const float* Wih_e  = (const float*)d_in[4];
  const float* Whh_e  = (const float*)d_in[5];
  const float* bih_e  = (const float*)d_in[6];
  const float* bhh_e  = (const float*)d_in[7];
  const float* W1     = (const float*)d_in[8];
  const float* b1     = (const float*)d_in[9];
  const float* W2     = (const float*)d_in[10];
  const float* b2     = (const float*)d_in[11];
  const float* Wih_d  = (const float*)d_in[12];
  const float* Whh_d  = (const float*)d_in[13];
  const float* bih_d  = (const float*)d_in[14];
  const float* bhh_d  = (const float*)d_in[15];
  const float* fc_w   = (const float*)d_in[16];
  const float* fc_b   = (const float*)d_in[17];
  const float* ff_w   = (const float*)d_in[18];
  const float* ff_b   = (const float*)d_in[19];
  float* out = (float*)d_out;

  __bf16* ws = (__bf16*)d_ws;
  size_t o = 0;
  __bf16* wcat  = ws + o; o += (size_t)4096 * 1536;   // [Wih_e|Whh_e] packed
  __bf16* whhd  = ws + o; o += (size_t)4096 * 1024;   // Whh_d packed
  __bf16* w1dc  = ws + o; o += (size_t)1024 * 2048;   // W1[:, :2048] packed
  __bf16* w1x   = ws + o; o += (size_t)1024 * 1024;   // W1[:, 2048:] packed
  __bf16* xenc  = ws + o; o += (size_t)2048 * 32 * 1024;
  __bf16* xproj = ws + o;

  pack_b_cat<<<2048, 256, 0, stream>>>(Wih_e, 512, Whh_e, 1024, wcat, 4096);
  pack_b<<<2048, 256, 0, stream>>>(Whh_d, whhd, 4096, 1024, 1024, 0);
  pack_b<<<1024, 256, 0, stream>>>(W1, w1dc, 1024, 2048, 3072, 0);
  pack_b<<<512, 256, 0, stream>>>(W1, w1x, 1024, 1024, 3072, 2048);

  size_t smem = (size_t)16 * LD_ACT * 2 + (size_t)3 * 16 * LD_F * 4;  // 263168 B
  da_rnn_main<<<128, 256, smem, stream>>>(X, y_prev, enc_w, bih_e, bhh_e, b1, W2, b2,
                                          Wih_d, bih_d, bhh_d, fc_w, fc_b, ff_w, ff_b,
                                          wcat, whhd, w1dc, w1x, xenc, xproj, out);
}
